// RSNNet_6511170421300
// MI455X (gfx1250) — matrix operand order repaired and checked by reading, not compiled, not run on hardware
//
#include <hip/hip_runtime.h>

typedef __attribute__((ext_vector_type(16))) __bf16 v16bf;
typedef __attribute__((ext_vector_type(8)))  __bf16 v8bf;
typedef __attribute__((ext_vector_type(8)))  float  v8f;
typedef __attribute__((ext_vector_type(4)))  int    v4i;

#define T_STEPS 500
#define B_SZ    256
#define C_SZ    700
#define CP_SZ   704   // C padded to multiple of 32 (zero-filled)
#define N_SZ    512
#define O_SZ    20
#define OP_SZ   32    // O padded to 2 WMMA tiles (zero rows)

// ---------------------------------------------------------------------------
// CDNA5 async global->LDS staging (ASYNCcnt path). Guarded so a toolchain
// without the builtin still compiles (plain copy fallback).
// Builtin signature (from compiler diagnostic): first arg is a *global*
// (AS1) int4 pointer, second the LDS (AS3) int4 pointer, then imm offset/cpol.
// ---------------------------------------------------------------------------
#if defined(__HIP_DEVICE_COMPILE__) && \
    __has_builtin(__builtin_amdgcn_global_load_async_to_lds_b128)
#define HAVE_ASYNC_LDS 1
#endif

__device__ __forceinline__ void async_copy_to_lds(const void* gsrc, void* ldst,
                                                  int bytes, int tid, int nthreads) {
#ifdef HAVE_ASYNC_LDS
  typedef __attribute__((address_space(1))) v4i* gptr_t;
  typedef __attribute__((address_space(3))) v4i* lptr_t;
  gptr_t g = (gptr_t)gsrc;
  lptr_t l = (lptr_t)ldst;
  const int n16 = bytes >> 4;
  for (int i = tid; i < n16; i += nthreads)
    __builtin_amdgcn_global_load_async_to_lds_b128(g + i, l + i, 0, 0);
#else
  const uint4* g = (const uint4*)gsrc;
  uint4* l = (uint4*)ldst;
  const int n16 = bytes >> 4;
  for (int i = tid; i < n16; i += nthreads) l[i] = g[i];
#endif
}

__device__ __forceinline__ void wait_async_lds() {
#ifdef HAVE_ASYNC_LDS
#if __has_builtin(__builtin_amdgcn_s_wait_asynccnt)
  __builtin_amdgcn_s_wait_asynccnt(0);
#else
  asm volatile("s_wait_asynccnt 0" ::: "memory");
#endif
#endif
}

// ---------------------------------------------------------------------------
// WMMA fragment helpers (CDNA5 16x16x32 bf16, wave32). Layouts per ISA 7.12.2:
//   A (16x32, 16-bit): lane L holds row M=L&15; K chunks {0..7,16..23} for
//     lanes 0-15 and {8..15,24..31} for lanes 16-31.
//   B (32x16, 16-bit): lane L holds col N=L&15; K chunks {0..7,16..23} for
//     lanes 0-15 and {8..15,24..31} for lanes 16-31 (same as A).
//   C/D (16x16 f32):   lane L holds col N=L&15; VGPR v holds row (L>>4)*8+v.
// ---------------------------------------------------------------------------
__device__ __forceinline__ v16bf load_a_frag(const __bf16* base, int ld, int k0, int lane) {
  const int m  = lane & 15;
  const int hi = lane >> 4;                    // 0 or 1
  const __bf16* p = base + m * ld + k0 + hi * 8;
  v8bf lo = *(const v8bf*)(p);
  v8bf hh = *(const v8bf*)(p + 16);
  v16bf a;
#pragma unroll
  for (int i = 0; i < 8; ++i) { a[i] = lo[i]; a[i + 8] = hh[i]; }
  return a;
}

// B[k][n] = W[n][k]  (W row-major [N][K]) -> two 16B loads per lane (K chunks as A).
__device__ __forceinline__ v16bf load_b_frag(const __bf16* w, int ldk, int n0, int k0, int lane) {
  const int n    = n0 + (lane & 15);
  const int koff = k0 + (lane >> 4) * 8;
  const __bf16* p = w + (size_t)n * ldk + koff;
  v8bf lo = *(const v8bf*)(p);
  v8bf hh = *(const v8bf*)(p + 16);
  v16bf b;
#pragma unroll
  for (int i = 0; i < 8; ++i) { b[i] = lo[i]; b[i + 8] = hh[i]; }
  return b;
}

__device__ __forceinline__ v8f wmma_bf16(v16bf a, v16bf b, v8f c) {
  return __builtin_amdgcn_wmma_f32_16x16x32_bf16(false, a, false, b, (short)0, c, false, false);
}

// ---------------------------------------------------------------------------
// Kernel 0: transpose/convert x [B][C][T] f32 -> Xt [T][B][CP] bf16 (zero pad)
// ---------------------------------------------------------------------------
__global__ void xt_transpose_kernel(const float* __restrict__ x, __bf16* __restrict__ Xt) {
  size_t i = (size_t)blockIdx.x * blockDim.x + threadIdx.x;
  const size_t total = (size_t)T_STEPS * B_SZ * CP_SZ;
  if (i >= total) return;
  int c = (int)(i % CP_SZ);
  int b = (int)((i / CP_SZ) % B_SZ);
  int t = (int)(i / ((size_t)CP_SZ * B_SZ));
  float v = 0.0f;
  if (c < C_SZ) v = x[(size_t)b * C_SZ * T_STEPS + (size_t)c * T_STEPS + t];
  Xt[i] = (__bf16)v;
}

// ---------------------------------------------------------------------------
// Kernel 1: f32 -> bf16 weight convert with zero padding
// ---------------------------------------------------------------------------
__global__ void convert_pad_kernel(const float* __restrict__ src, __bf16* __restrict__ dst,
                                   int src_rows, int src_cols, int dst_rows, int dst_cols) {
  int i = blockIdx.x * blockDim.x + threadIdx.x;
  int total = dst_rows * dst_cols;
  if (i >= total) return;
  int r = i / dst_cols, c = i - r * dst_cols;
  float v = (r < src_rows && c < src_cols) ? src[r * src_cols + c] : 0.0f;
  dst[i] = (__bf16)v;
}

// ---------------------------------------------------------------------------
// Kernel 2: H1[t,b,n] = x[b,:,t] @ W1^T + b1[n] + br2[n]   (bf16 out)
// grid = T * (B/16) * (N/256), block = 128 (4 waves x 4 n-tiles each).
// The 16x704 A-tile is staged once into LDS (async) and shared by all waves:
// 4 WMMAs per A fragment fetch.
// ---------------------------------------------------------------------------
__global__ void __launch_bounds__(128) h1_gemm_kernel(const __bf16* __restrict__ Xt,
                                                      const __bf16* __restrict__ W1p,
                                                      const float* __restrict__ b1,
                                                      const float* __restrict__ br2,
                                                      __bf16* __restrict__ H1) {
  __shared__ __bf16 atile[16 * CP_SZ];          // 22.5 KB
  const int lane = threadIdx.x & 31;
  const int wave = threadIdx.x >> 5;
  int blk = blockIdx.x;
  const int NG = N_SZ / 256;                    // 2
  int ng = blk % NG;              blk /= NG;
  int bb = blk % (B_SZ / 16);     int t = blk / (B_SZ / 16);
  const int nb    = ng * 256 + wave * 64;       // wave covers nb .. nb+63
  const int brow0 = bb * 16;

  const __bf16* abase = Xt + (size_t)t * B_SZ * CP_SZ + (size_t)brow0 * CP_SZ;
  async_copy_to_lds(abase, atile, 16 * CP_SZ * 2, threadIdx.x, 128);
  wait_async_lds();
  __syncthreads();

  v8f acc[4] = {{}, {}, {}, {}};
  for (int k0 = 0; k0 < CP_SZ; k0 += 32) {
    v16bf a = load_a_frag(atile, CP_SZ, k0, lane);
#pragma unroll
    for (int i = 0; i < 4; ++i) {
      v16bf b = load_b_frag(W1p, CP_SZ, nb + i * 16, k0, lane);
      acc[i] = wmma_bf16(a, b, acc[i]);
    }
  }

  __bf16* obase = H1 + (size_t)t * B_SZ * N_SZ + (size_t)brow0 * N_SZ;
  const int mh = (lane >> 4) * 8;
#pragma unroll
  for (int i = 0; i < 4; ++i) {
    const int n = nb + i * 16 + (lane & 15);
    const float bias = b1[n] + br2[n];
#pragma unroll
    for (int v = 0; v < 8; ++v)
      obase[(size_t)(mh + v) * N_SZ + n] = (__bf16)(acc[i][v] + bias);
  }
}

// ---------------------------------------------------------------------------
// Recurrent scan helpers
// ---------------------------------------------------------------------------
__device__ __forceinline__ void gemm2_to_hb(const __bf16* sA, const __bf16* WA,
                                            const __bf16* sB, const __bf16* WB,
                                            float* hb, int wave, int lane) {
  v8f acc[4] = {{}, {}, {}, {}};
  for (int k0 = 0; k0 < N_SZ; k0 += 32) {
    v16bf a = load_a_frag(sA, N_SZ, k0, lane);
#pragma unroll
    for (int i = 0; i < 4; ++i) {
      v16bf b = load_b_frag(WA, N_SZ, (wave * 4 + i) * 16, k0, lane);
      acc[i] = wmma_bf16(a, b, acc[i]);
    }
  }
  for (int k0 = 0; k0 < N_SZ; k0 += 32) {
    v16bf a = load_a_frag(sB, N_SZ, k0, lane);
#pragma unroll
    for (int i = 0; i < 4; ++i) {
      v16bf b = load_b_frag(WB, N_SZ, (wave * 4 + i) * 16, k0, lane);
      acc[i] = wmma_bf16(a, b, acc[i]);
    }
  }
  const int mh = (lane >> 4) * 8;
#pragma unroll
  for (int i = 0; i < 4; ++i) {
    int n = (wave * 4 + i) * 16 + (lane & 15);
#pragma unroll
    for (int v = 0; v < 8; ++v)
      hb[(mh + v) * N_SZ + n] = acc[i][v];
  }
}

__device__ __forceinline__ void lif_update(float* mA, unsigned short* cA, __bf16* sA,
                                           const float* hb,
                                           const float* tau, const float* vth,
                                           const float* bA, const float* bB,
                                           int tid) {
  for (int e = tid; e < 16 * N_SZ; e += 256) {
    int n = e & (N_SZ - 1);
    float h = hb[e];
    if (bA) h += bA[n] + bB[n];
    float mo = mA[e];
    float so = (float)sA[e];
    float mn = tau[n] * mo * (1.0f - so) + h;
    int sp = (mn - vth[n]) > 0.0f ? 1 : 0;
    mA[e] = mn;
    cA[e] = (unsigned short)(cA[e] + sp);
    sA[e] = (__bf16)(float)sp;
  }
}

// ---------------------------------------------------------------------------
// Kernel 3: recurrent scan. 16 blocks x 256 threads (8 waves); each block owns
// a 16-row batch tile. All LIF state is LDS-resident (~272 KB of 320 KB WGP):
//   spikes bf16, membranes f32, spike counters u16, h-buffer f32,
//   async-staged H1[t] tile, LDS-resident padded W4.
// H1[t] is async-loaded into LDS while the layer-1 recurrent WMMA runs.
// ---------------------------------------------------------------------------
__global__ void __launch_bounds__(256) rsnn_recurrent_kernel(
    const __bf16* __restrict__ H1,
    const __bf16* __restrict__ W2b, const __bf16* __restrict__ Wr2b,
    const __bf16* __restrict__ W3b, const __bf16* __restrict__ Wr3b,
    const __bf16* __restrict__ W4p,
    const float* __restrict__ b2, const float* __restrict__ br2,
    const float* __restrict__ b3, const float* __restrict__ br3,
    const float* __restrict__ b4,
    const float* __restrict__ tau1, const float* __restrict__ Vth1,
    const float* __restrict__ tau2, const float* __restrict__ Vth2,
    const float* __restrict__ tau3, const float* __restrict__ Vth3,
    float* __restrict__ out, float* __restrict__ sc_out) {
  extern __shared__ char smem[];
  const int TILE = 16 * N_SZ;                 // 8192 elements
  __bf16*         s1    = (__bf16*)smem;                     // 16 KB
  __bf16*         s2    = s1 + TILE;                         // 16 KB
  __bf16*         s3    = s2 + TILE;                         // 16 KB
  float*          m1    = (float*)(s3 + TILE);               // 32 KB
  float*          m2    = m1 + TILE;                         // 32 KB
  float*          m3    = m2 + TILE;                         // 32 KB
  unsigned short* c1    = (unsigned short*)(m3 + TILE);      // 16 KB
  unsigned short* c2    = c1 + TILE;                         // 16 KB
  unsigned short* c3    = c2 + TILE;                         // 16 KB
  float*          hb    = (float*)(c3 + TILE);               // 32 KB
  __bf16*         h1lds = (__bf16*)(hb + TILE);              // 16 KB
  __bf16*         w4lds = h1lds + TILE;                      // 32 KB -> 272 KB total

  const int tid   = threadIdx.x;
  const int lane  = tid & 31;
  const int wave  = tid >> 5;
  const int brow0 = blockIdx.x * 16;

  // Stage W4 (padded 32x512 bf16) into LDS once; used 500x by waves 0-1.
  async_copy_to_lds(W4p, w4lds, OP_SZ * N_SZ * 2, tid, 256);
  for (int e = tid; e < TILE; e += 256) {
    s1[e] = (__bf16)0.0f; s2[e] = (__bf16)0.0f; s3[e] = (__bf16)0.0f;
    m1[e] = 0.0f; m2[e] = 0.0f; m3[e] = 0.0f;
    c1[e] = 0; c2[e] = 0; c3[e] = 0;
  }
  wait_async_lds();
  __syncthreads();

  for (int t = 0; t < T_STEPS; ++t) {
    // Kick off the async H1[t] tile fetch; it lands while GEMM1 runs.
    const __bf16* h1row = H1 + (size_t)t * B_SZ * N_SZ + (size_t)brow0 * N_SZ;
    async_copy_to_lds(h1row, h1lds, TILE * 2, tid, 256);
    if (t + 1 < T_STEPS)  // warm L2 for the next step's feed-forward slice
      __builtin_prefetch(h1row + (size_t)B_SZ * N_SZ + (size_t)wave * 1024, 0, 1);

    // ---- layer 1: h1 = H1[t] + s1 @ Wr2^T  (b1+br2 already folded into H1)
    {
      v8f acc[4] = {{}, {}, {}, {}};
      for (int k0 = 0; k0 < N_SZ; k0 += 32) {
        v16bf a = load_a_frag(s1, N_SZ, k0, lane);
#pragma unroll
        for (int i = 0; i < 4; ++i) {
          v16bf b = load_b_frag(Wr2b, N_SZ, (wave * 4 + i) * 16, k0, lane);
          acc[i] = wmma_bf16(a, b, acc[i]);
        }
      }
      wait_async_lds();
      __syncthreads();                 // all waves' async H1 chunks visible
      const int mh = (lane >> 4) * 8;
#pragma unroll
      for (int i = 0; i < 4; ++i) {
        int n = (wave * 4 + i) * 16 + (lane & 15);
#pragma unroll
        for (int v = 0; v < 8; ++v)
          hb[(mh + v) * N_SZ + n] = acc[i][v] + (float)h1lds[(mh + v) * N_SZ + n];
      }
    }
    __syncthreads();
    lif_update(m1, c1, s1, hb, tau1, Vth1, nullptr, nullptr, tid);
    __syncthreads();

    // ---- layer 2: h2 = s1_new @ W2^T + s2_old @ Wr2^T  (+ b2+br2 in update)
    gemm2_to_hb(s1, W2b, s2, Wr2b, hb, wave, lane);
    __syncthreads();
    lif_update(m2, c2, s2, hb, tau2, Vth2, b2, br2, tid);
    __syncthreads();

    // ---- layer 3: h3 = s2_new @ W3^T + s3_old @ Wr3^T  (+ b3+br3 in update)
    gemm2_to_hb(s2, W3b, s3, Wr3b, hb, wave, lane);
    __syncthreads();
    lif_update(m3, c3, s3, hb, tau3, Vth3, b3, br3, tid);
    __syncthreads();

    // ---- output: o = s3_new @ W4^T + b4 ; W4 padded to 32 rows -> 2 tiles
    if (wave < 2) {
      int n0 = wave * 16;
      v8f acc = {};
      for (int k0 = 0; k0 < N_SZ; k0 += 32) {
        v16bf a = load_a_frag(s3, N_SZ, k0, lane);
        v16bf b = load_b_frag(w4lds, N_SZ, n0, k0, lane);
        acc = wmma_bf16(a, b, acc);
      }
      int n = n0 + (lane & 15);
      const int mh = (lane >> 4) * 8;
      if (n < O_SZ) {
#pragma unroll
        for (int v = 0; v < 8; ++v)
          out[(size_t)(brow0 + mh + v) * O_SZ * T_STEPS + (size_t)n * T_STEPS + t] =
              acc[v] + b4[n];
      }
    }
    __syncthreads();
  }

  const float invT = 1.0f / (float)T_STEPS;
  for (int e = tid; e < TILE; e += 256) {
    int b = e >> 9;
    int n = e & (N_SZ - 1);
    size_t g = (size_t)(brow0 + b) * N_SZ + n;
    sc_out[g]                           = (float)c1[e] * invT;
    sc_out[(size_t)B_SZ * N_SZ + g]     = (float)c2[e] * invT;
    sc_out[2 * (size_t)B_SZ * N_SZ + g] = (float)c3[e] * invT;
  }
}

// ---------------------------------------------------------------------------
// Host launcher
// ---------------------------------------------------------------------------
extern "C" void kernel_launch(void* const* d_in, const int* in_sizes, int n_in,
                              void* d_out, int out_size, void* d_ws, size_t ws_size,
                              hipStream_t stream) {
  (void)in_sizes; (void)n_in; (void)out_size; (void)ws_size;
  const float* x    = (const float*)d_in[0];
  const float* W1   = (const float*)d_in[1];
  const float* b1   = (const float*)d_in[2];
  const float* W2   = (const float*)d_in[3];
  const float* b2   = (const float*)d_in[4];
  const float* Wr2  = (const float*)d_in[5];
  const float* br2  = (const float*)d_in[6];
  const float* W3   = (const float*)d_in[7];
  const float* b3   = (const float*)d_in[8];
  const float* Wr3  = (const float*)d_in[9];
  const float* br3  = (const float*)d_in[10];
  const float* W4   = (const float*)d_in[11];
  const float* b4   = (const float*)d_in[12];
  const float* Vth1 = (const float*)d_in[13];
  const float* tau1 = (const float*)d_in[14];
  const float* Vth2 = (const float*)d_in[15];
  const float* tau2 = (const float*)d_in[16];
  const float* Vth3 = (const float*)d_in[17];
  const float* tau3 = (const float*)d_in[18];

  float* out    = (float*)d_out;
  float* sc_out = out + (size_t)B_SZ * O_SZ * T_STEPS;

  size_t off = 0;
  auto carve = [&](size_t bytes) -> char* {
    char* p = (char*)d_ws + off;
    off += (bytes + 255) & ~(size_t)255;
    return p;
  };
  __bf16* Xt   = (__bf16*)carve((size_t)T_STEPS * B_SZ * CP_SZ * 2);
  __bf16* H1   = (__bf16*)carve((size_t)T_STEPS * B_SZ * N_SZ * 2);
  __bf16* W1p  = (__bf16*)carve((size_t)N_SZ * CP_SZ * 2);
  __bf16* W2b  = (__bf16*)carve((size_t)N_SZ * N_SZ * 2);
  __bf16* Wr2b = (__bf16*)carve((size_t)N_SZ * N_SZ * 2);
  __bf16* W3b  = (__bf16*)carve((size_t)N_SZ * N_SZ * 2);
  __bf16* Wr3b = (__bf16*)carve((size_t)N_SZ * N_SZ * 2);
  __bf16* W4p  = (__bf16*)carve((size_t)OP_SZ * N_SZ * 2);

  {
    size_t total = (size_t)T_STEPS * B_SZ * CP_SZ;
    int blocks = (int)((total + 255) / 256);
    xt_transpose_kernel<<<blocks, 256, 0, stream>>>(x, Xt);
  }
  auto convert = [&](const float* src, __bf16* dst, int sr, int sc, int dr, int dc) {
    int total = dr * dc;
    convert_pad_kernel<<<(total + 255) / 256, 256, 0, stream>>>(src, dst, sr, sc, dr, dc);
  };
  convert(W1,  W1p,  N_SZ, C_SZ, N_SZ, CP_SZ);
  convert(W2,  W2b,  N_SZ, N_SZ, N_SZ, N_SZ);
  convert(Wr2, Wr2b, N_SZ, N_SZ, N_SZ, N_SZ);
  convert(W3,  W3b,  N_SZ, N_SZ, N_SZ, N_SZ);
  convert(Wr3, Wr3b, N_SZ, N_SZ, N_SZ, N_SZ);
  convert(W4,  W4p,  O_SZ, N_SZ, OP_SZ, N_SZ);

  {
    int blocks = T_STEPS * (B_SZ / 16) * (N_SZ / 256);
    h1_gemm_kernel<<<blocks, 128, 0, stream>>>(Xt, W1p, b1, br2, H1);
  }
  {
    size_t shmem = (size_t)(3 * 16 * N_SZ) * 2     // s1..s3 bf16
                 + (size_t)(4 * 16 * N_SZ) * 4     // m1..3, hb (f32)
                 + (size_t)(3 * 16 * N_SZ) * 2     // c1..3 (u16)
                 + (size_t)(16 * N_SZ) * 2         // h1 staging tile (bf16)
                 + (size_t)(OP_SZ * N_SZ) * 2;     // W4 in LDS (bf16)
    (void)hipFuncSetAttribute((const void*)rsnn_recurrent_kernel,
                              hipFuncAttributeMaxDynamicSharedMemorySize, (int)shmem);
    rsnn_recurrent_kernel<<<B_SZ / 16, 256, shmem, stream>>>(
        H1, W2b, Wr2b, W3b, Wr3b, W4p,
        b2, br2, b3, br3, b4,
        tau1, Vth1, tau2, Vth2, tau3, Vth3,
        out, sc_out);
  }
}
